// PhaseMoE_50938312131044
// MI455X (gfx1250) — matrix operand order repaired and checked by reading, not compiled, not run on hardware
//
#include <hip/hip_runtime.h>

// ---------------- CDNA5 WMMA types ----------------
typedef __attribute__((ext_vector_type(16))) __bf16 v16bf;
typedef __attribute__((ext_vector_type(8)))  float  v8f;

#define BTOK   32768
#define KEXP   8
#define LATENT 256
#define CONDD  1024
#define HID    128
#define D0     1280      // u(256) + cond(1024); tau folded into bias
#define MTILE  64        // tokens per block
#define XPAD   1288      // padded row stride (bf16) for x tile  (breaks bank conflicts)
#define HPAD   136       // padded row stride (bf16) for h tiles

union FragCast { struct { uint4 a, b; } u; v16bf v; };

__device__ __forceinline__ unsigned short f2bf(float x) {
  unsigned uu = __float_as_uint(x);
  unsigned r  = uu + 0x7FFFu + ((uu >> 16) & 1u);   // RNE
  return (unsigned short)(r >> 16);
}
__device__ __forceinline__ unsigned pack2(float lo, float hi) {
  return (unsigned)f2bf(lo) | ((unsigned)f2bf(hi) << 16);
}
__device__ __forceinline__ float silu(float x) { return x / (1.0f + __expf(-x)); }

// B fragment: lane holds 16 contiguous K values (fragment-major layout in d_ws)
__device__ __forceinline__ v16bf load_b_frag(const __bf16* p) {
  FragCast f;
  f.u.a = *(const uint4*)(p);
  f.u.b = *(const uint4*)(p + 8);
  return f.v;
}
// A fragment from LDS: lane = (m=ln, kh): VGPR0-3 hold K[kh*16..+7],
// VGPR4-7 hold K[kh*16+8..+7]  (two 16B chunks)
__device__ __forceinline__ v16bf load_a_frag(const __bf16* row, int colBase, int kh) {
  FragCast f;
  f.u.a = *(const uint4*)(row + colBase + kh * 16);
  f.u.b = *(const uint4*)(row + colBase + kh * 16 + 8);
  return f.v;
}

// ---------------- weight conversion pre-kernels (f32 -> bf16 fragment-major) ----------------
// W1F[k][nt=0..7][kc=0..39][lane=0..31][i=0..15] ; value = W1[k][d][n]
//   n = nt*16 + (lane&15) ; d = kc*32 + (lane>>4)*16 + i
__global__ void convW1(const float* __restrict__ W1, unsigned short* __restrict__ W1F) {
  int idx = blockIdx.x * 256 + threadIdx.x;           // < 1310720
  int i    = idx & 15;
  int lane = (idx >> 4) & 31;
  int kc   = (idx >> 9) % 40;
  int nt   = (idx / (512 * 40)) & 7;
  int k    =  idx / (512 * 40 * 8);
  int n = nt * 16 + (lane & 15);
  int d = kc * 32 + ((lane >> 4) << 4) + i;
  W1F[idx] = f2bf(W1[((size_t)k * 1281 + d) * HID + n]);
}
// W2F[k][nt=0..7][kc=0..3][lane][i] ; value = W2[k][d][n]
__global__ void convW2(const float* __restrict__ W2, unsigned short* __restrict__ W2F) {
  int idx = blockIdx.x * 256 + threadIdx.x;           // < 131072
  int i    = idx & 15;
  int lane = (idx >> 4) & 31;
  int kc   = (idx >> 9) & 3;
  int nt   = (idx >> 11) & 7;
  int k    =  idx >> 14;
  int n = nt * 16 + (lane & 15);
  int d = kc * 32 + ((lane >> 4) << 4) + i;
  W2F[idx] = f2bf(W2[((size_t)k * HID + d) * HID + n]);
}
// W3F[k][nt=0..15][kc=0..3][lane][i] ; value = W3[k][g][n] (n in 0..255)
__global__ void convW3(const float* __restrict__ W3, unsigned short* __restrict__ W3F) {
  int idx = blockIdx.x * 256 + threadIdx.x;           // < 262144
  int i    = idx & 15;
  int lane = (idx >> 4) & 31;
  int kc   = (idx >> 9) & 3;
  int nt   = (idx >> 11) & 15;
  int k    =  idx >> 15;
  int n = nt * 16 + (lane & 15);
  int g = kc * 32 + ((lane >> 4) << 4) + i;
  W3F[idx] = f2bf(W3[((size_t)k * HID + g) * LATENT + n]);
}
// wtau[k][n] = W1[k][1280][n]  (kept f32; folded into layer-1 bias with tau)
__global__ void convWT(const float* __restrict__ W1, float* __restrict__ wtau) {
  int idx = blockIdx.x * 256 + threadIdx.x;           // < 1024
  if (idx < KEXP * HID) {
    int k = idx / HID, n = idx % HID;
    wtau[idx] = W1[((size_t)k * 1281 + 1280) * HID + n];
  }
}

// ---------------- fused MoE kernel ----------------
// 512 threads = 16 waves: wn = wave&7 -> HID column tile, wm = wave>>3 -> 32-row half.
// 4 waves per SIMD32 for latency hiding; per-wave regs ~halved vs 8-wave version.
__global__ __launch_bounds__(512) void moe_fused(
    const float* __restrict__ cond, const float* __restrict__ u,
    const float* __restrict__ tau,  const float* __restrict__ p_hat,
    const float* __restrict__ b1,   const float* __restrict__ b2,
    const float* __restrict__ b3,
    const __bf16* __restrict__ W1F, const __bf16* __restrict__ W2F,
    const __bf16* __restrict__ W3F, const float* __restrict__ wtau,
    float* __restrict__ out)
{
  __shared__ __bf16 sX [MTILE * XPAD];   // 164864 B
  __shared__ __bf16 sH1[MTILE * HPAD];   //  17408 B
  __shared__ __bf16 sH2[MTILE * HPAD];   //  17408 B
  __shared__ float  sTau[MTILE];
  __shared__ float  sP  [MTILE * KEXP];

  const int tid  = threadIdx.x;
  const int wv   = tid >> 5;      // wave 0..15
  const int wn   = wv & 7;        // column tile of HID: cols [wn*16, wn*16+16)
  const int wm   = wv >> 3;       // row half: rows [wm*32, wm*32+32)
  const int lane = tid & 31;
  const int ln   = lane & 15;
  const int kh   = lane >> 4;
  const int row0 = blockIdx.x * MTILE;

  // ---- stage x tile (u|cond) as bf16 into LDS; 40 float4 chunks per thread ----
  #pragma unroll 4
  for (int it = 0; it < 40; ++it) {
    int chunk = tid + it * 512;                  // 0..20479 ; 320 chunks per row
    int r = chunk / 320;
    int c = (chunk - r * 320) * 4;
    const float* src = (c < 256) ? (u    + (size_t)(row0 + r) * 256  + c)
                                 : (cond + (size_t)(row0 + r) * 1024 + (c - 256));
    float4 v = *(const float4*)src;
    uint2 pk; pk.x = pack2(v.x, v.y); pk.y = pack2(v.z, v.w);
    *(uint2*)(&sX[r * XPAD + c]) = pk;
  }
  if (tid < MTILE) sTau[tid] = tau[row0 + tid];
  if (tid < MTILE * KEXP) sP[tid] = p_hat[(size_t)row0 * KEXP + tid];
  __syncthreads();

  // persistent p-weighted output accumulators: 2 M-tiles x 2 N-tiles of LATENT
  v8f accO[2][2];
  #pragma unroll
  for (int mt = 0; mt < 2; ++mt)
    #pragma unroll
    for (int n2 = 0; n2 < 2; ++n2) accO[mt][n2] = (v8f)(0.0f);

  const int ncol = wn * 16 + ln;    // this wave's HID column for this lane

  for (int k = 0; k < KEXP; ++k) {
    // ================= layer 1 : (64 x 1280) @ (1280 x 128) =================
    v8f acc1[2];
    #pragma unroll
    for (int mt = 0; mt < 2; ++mt) acc1[mt] = (v8f)(0.0f);
    const __bf16* b1base = W1F + ((size_t)(k * 8 + wn) * 40 * 32 + lane) * 16;
    const __bf16* b1next = W1F + ((size_t)((((k + 1) & 7) * 8) + wn) * 40 * 32 + lane) * 16;
    for (int kc = 0; kc < 40; ++kc) {
      __builtin_prefetch(b1next + (size_t)kc * 512, 0, 2);   // warm next expert's W1 stream
      v16bf bfrag = load_b_frag(b1base + (size_t)kc * 512);
      #pragma unroll
      for (int mt = 0; mt < 2; ++mt) {
        v16bf afrag = load_a_frag(&sX[(wm * 32 + mt * 16 + ln) * XPAD], kc * 32, kh);
        acc1[mt] = __builtin_amdgcn_wmma_f32_16x16x32_bf16(
            false, afrag, false, bfrag, (short)0, acc1[mt], false, false);
      }
    }
    { // epilogue: + b1 + tau*W1[:,1280,:], silu, -> sH1 (bf16)
      float b1v = b1[k * HID + ncol];
      float wtv = wtau[k * HID + ncol];
      unsigned short* h1s = (unsigned short*)sH1;
      #pragma unroll
      for (int mt = 0; mt < 2; ++mt)
        #pragma unroll
        for (int j = 0; j < 8; ++j) {
          int m = wm * 32 + mt * 16 + j + kh * 8;
          float pre = acc1[mt][j] + b1v + sTau[m] * wtv;
          h1s[m * HPAD + ncol] = f2bf(silu(pre));
        }
    }
    __syncthreads();

    // ================= layer 2 : (64 x 128) @ (128 x 128) =================
    v8f acc2[2];
    #pragma unroll
    for (int mt = 0; mt < 2; ++mt) acc2[mt] = (v8f)(0.0f);
    const __bf16* b2base = W2F + ((size_t)(k * 8 + wn) * 4 * 32 + lane) * 16;
    #pragma unroll
    for (int kc = 0; kc < 4; ++kc) {
      v16bf bfrag = load_b_frag(b2base + kc * 512);
      #pragma unroll
      for (int mt = 0; mt < 2; ++mt) {
        v16bf afrag = load_a_frag(&sH1[(wm * 32 + mt * 16 + ln) * HPAD], kc * 32, kh);
        acc2[mt] = __builtin_amdgcn_wmma_f32_16x16x32_bf16(
            false, afrag, false, bfrag, (short)0, acc2[mt], false, false);
      }
    }
    { // epilogue: + b2, silu, -> sH2 (bf16)
      float b2v = b2[k * HID + ncol];
      unsigned short* h2s = (unsigned short*)sH2;
      #pragma unroll
      for (int mt = 0; mt < 2; ++mt)
        #pragma unroll
        for (int j = 0; j < 8; ++j) {
          int m = wm * 32 + mt * 16 + j + kh * 8;
          h2s[m * HPAD + ncol] = f2bf(silu(acc2[mt][j] + b2v));
        }
    }
    __syncthreads();

    // ========= layer 3 : (64 x 128) @ (128 x 256), p-weighted accumulate =========
    #pragma unroll
    for (int n2 = 0; n2 < 2; ++n2) {
      int nt3 = wn * 2 + n2;                     // N-tile of LATENT (0..15)
      v8f acc3[2];
      #pragma unroll
      for (int mt = 0; mt < 2; ++mt) acc3[mt] = (v8f)(0.0f);
      const __bf16* b3base = W3F + ((size_t)(k * 16 + nt3) * 4 * 32 + lane) * 16;
      #pragma unroll
      for (int kc = 0; kc < 4; ++kc) {
        v16bf bfrag = load_b_frag(b3base + kc * 512);
        #pragma unroll
        for (int mt = 0; mt < 2; ++mt) {
          v16bf afrag = load_a_frag(&sH2[(wm * 32 + mt * 16 + ln) * HPAD], kc * 32, kh);
          acc3[mt] = __builtin_amdgcn_wmma_f32_16x16x32_bf16(
              false, afrag, false, bfrag, (short)0, acc3[mt], false, false);
        }
      }
      int ncol3 = nt3 * 16 + ln;
      float b3v = b3[k * LATENT + ncol3];
      #pragma unroll
      for (int mt = 0; mt < 2; ++mt)
        #pragma unroll
        for (int j = 0; j < 8; ++j) {
          int m = wm * 32 + mt * 16 + j + kh * 8;
          accO[mt][n2][j] += sP[m * KEXP + k] * (acc3[mt][j] + b3v);
        }
    }
  } // expert loop

  // ---- write out (B, 256) f32 ----
  #pragma unroll
  for (int n2 = 0; n2 < 2; ++n2) {
    int colb = (wn * 2 + n2) * 16 + ln;
    #pragma unroll
    for (int mt = 0; mt < 2; ++mt)
      #pragma unroll
      for (int j = 0; j < 8; ++j) {
        int row = row0 + wm * 32 + mt * 16 + j + kh * 8;
        out[(size_t)row * LATENT + colb] = accO[mt][n2][j];
      }
  }
}

// ---------------- host launcher ----------------
extern "C" void kernel_launch(void* const* d_in, const int* in_sizes, int n_in,
                              void* d_out, int out_size, void* d_ws, size_t ws_size,
                              hipStream_t stream) {
  (void)in_sizes; (void)n_in; (void)out_size; (void)ws_size;
  const float* cond = (const float*)d_in[0];
  const float* u    = (const float*)d_in[1];
  const float* tau  = (const float*)d_in[2];
  const float* ph   = (const float*)d_in[3];
  const float* W1   = (const float*)d_in[4];
  const float* b1   = (const float*)d_in[5];
  const float* W2   = (const float*)d_in[6];
  const float* b2   = (const float*)d_in[7];
  const float* W3   = (const float*)d_in[8];
  const float* b3   = (const float*)d_in[9];

  // workspace layout (bf16 fragment-major weights + f32 tau-row) ~3.4 MB
  unsigned short* W1F = (unsigned short*)d_ws;          // 1,310,720 elems
  unsigned short* W2F = W1F + 1310720;                  //   131,072
  unsigned short* W3F = W2F + 131072;                   //   262,144
  float*          wtv = (float*)(W3F + 262144);         //     1,024

  convW1<<<5120, 256, 0, stream>>>(W1, W1F);
  convW2<<<512,  256, 0, stream>>>(W2, W2F);
  convW3<<<1024, 256, 0, stream>>>(W3, W3F);
  convWT<<<4,    256, 0, stream>>>(W1, wtv);

  moe_fused<<<BTOK / MTILE, 512, 0, stream>>>(
      cond, u, tau, ph, b1, b2, b3,
      (const __bf16*)W1F, (const __bf16*)W2F, (const __bf16*)W3F, wtv,
      (float*)d_out);
}
